// DecoderRealBenchmark_90477781057697
// MI455X (gfx1250) — compile-verified, matrix operand order checked by reading, not run on hardware
//
#include <hip/hip_runtime.h>
#include <hip/hip_bf16.h>

// ---------------------------------------------------------------------------
// Types for CDNA5 WMMA (wave32): v_wmma_f32_16x16x32_bf16
// ---------------------------------------------------------------------------
typedef __bf16 bf16_t;
typedef __attribute__((ext_vector_type(16))) __bf16 v16bf;
typedef __attribute__((ext_vector_type(8)))  __bf16 v8bf;
typedef __attribute__((ext_vector_type(8)))  float  v8f;

// Load a 16-element bf16 WMMA fragment for one lane from an LDS tile whose
// rows are k-contiguous. Per the ISA 16-bit A/B layout:
//   lanes 0-15 : elems 0-7 -> K=0..7,   elems 8-15 -> K=16..23
//   lanes 16-31: elems 0-7 -> K=8..15,  elems 8-15 -> K=24..31
// i.e. two contiguous 8-half (16B) chunks => two ds_load_b128.
__device__ __forceinline__ v16bf load_frag(const bf16_t* row_base, int kbase) {
  const v8bf lo = *reinterpret_cast<const v8bf*>(row_base + kbase);
  const v8bf hi = *reinterpret_cast<const v8bf*>(row_base + kbase + 16);
  v16bf f;
#pragma unroll
  for (int i = 0; i < 8; ++i) { f[i] = lo[i]; f[8 + i] = hi[i]; }
  return f;
}

__device__ __forceinline__ v8f wmma_bf16(v16bf a, v16bf b, v8f c) {
  // (neg_a, A, neg_b, B, c_mod, C, reuse_a, reuse_b)
  return __builtin_amdgcn_wmma_f32_16x16x32_bf16(false, a, false, b, (short)0, c,
                                                 false, false);
}

// gfx1250 async direct-to-LDS copy (VGPR-bypassing), tracked by ASYNCcnt.
// Generic shared pointers carry the LDS byte offset in their low 32 bits
// (flat aperture mapping).
__device__ __forceinline__ uint32_t lds_off32(const void* p) {
  return (uint32_t)(uintptr_t)p;
}
__device__ __forceinline__ void async_copy_b128(uint32_t lds_off,
                                                const void* gptr) {
  asm volatile("global_load_async_to_lds_b128 %0, %1, off"
               :: "v"(lds_off), "v"(gptr) : "memory");
}
__device__ __forceinline__ void wait_async0() {
  asm volatile("s_wait_asynccnt 0x0" ::: "memory");
}

__device__ __forceinline__ float sigmoidf_(float x) {
  return 1.0f / (1.0f + __expf(-x));
}

// ---------------------------------------------------------------------------
// Problem constants
// ---------------------------------------------------------------------------
#define BSZ 512   // batch
#define LSZ 512   // hidden
#define TSZ 100   // timesteps
#define KTOT 640  // 128 (input: [a_t | t/100]) + 512 (hidden)
#define NPAD 576  // head intermediate cols padded 513 -> 576 (18*32)

// ---------------------------------------------------------------------------
// Kernel 0: one-shot conversions (fp32 -> bf16 weights, init state)
// ---------------------------------------------------------------------------
__global__ void prep_kernel(const float* __restrict__ W_ih,
                            const float* __restrict__ W_hh,
                            const float* __restrict__ W1,
                            const float* __restrict__ W2,
                            const float* __restrict__ init_,
                            bf16_t* __restrict__ Wih_bf,
                            bf16_t* __restrict__ Whh_bf,
                            bf16_t* __restrict__ W1_bf,
                            bf16_t* __restrict__ W2_bf,
                            bf16_t* __restrict__ h0,
                            float*  __restrict__ cst) {
  long i = (long)blockIdx.x * blockDim.x + threadIdx.x;
  const long N0 = 2048L * 128;      // W_ih
  const long N1 = 2048L * 512;      // W_hh
  const long N2 = 513L * 512;       // W1
  const long N3 = 64L * NPAD;       // W2 padded to K=576
  const long N4 = (long)BSZ * LSZ;  // h0
  const long N5 = (long)BSZ * LSZ;  // c
  if (i < N0) { Wih_bf[i] = (bf16_t)W_ih[i]; return; } i -= N0;
  if (i < N1) { Whh_bf[i] = (bf16_t)W_hh[i]; return; } i -= N1;
  if (i < N2) { W1_bf[i] = (bf16_t)W1[i];    return; } i -= N2;
  if (i < N3) {
    long n = i / NPAD, k = i % NPAD;
    W2_bf[i] = (k < 513) ? (bf16_t)W2[n * 513 + k] : (bf16_t)0.0f;
    return;
  } i -= N3;
  if (i < N4) { h0[i] = (bf16_t)init_[i]; return; } i -= N4;
  if (i < N5) { cst[i] = init_[i]; }
}

// ---------------------------------------------------------------------------
// Kernel 1: one LSTM timestep, fully fused.
// grid = (LSZ/16, BSZ/32), block = 256 (8 waves).
// Wave w: gate = w&3 (torch order i,f,g,o), row-subtile = w>>2.
// K loop: 640 in tiles of 64 -> 2 WMMAs per wave per barrier pair.
// B-tiles (bf16 weights) staged with async direct-to-LDS copies;
// A-tile (needs fp32->bf16 convert) staged through the VALU.
// ---------------------------------------------------------------------------
__global__ __launch_bounds__(256) void lstm_step_kernel(
    const float*  __restrict__ a_t,      // (BSZ, 64)  slice at step t
    const bf16_t* __restrict__ hsrc,     // (BSZ, LSZ) bf16
    bf16_t*       __restrict__ hdst,     // (BSZ, LSZ) bf16
    float*        __restrict__ cstate,   // (BSZ, LSZ) fp32
    float*        __restrict__ hs_out_t, // (BSZ, LSZ) fp32 slice of d_out
    const bf16_t* __restrict__ Wih,      // (2048, 128)
    const bf16_t* __restrict__ Whh,      // (2048, 512)
    const float*  __restrict__ b_ih,     // 2048
    const float*  __restrict__ b_hh,     // 2048
    float tval)                          // t / 100
{
  __shared__ __align__(16) bf16_t As[32 * 64];      // batch-rows x K(64)
  __shared__ __align__(16) bf16_t Bs[4 * 16 * 64];  // [gate][n][k] k-contig
  __shared__ float Gs[4 * 32 * 16];                 // gate results [g][m][n]

  const int tid    = threadIdx.x;
  const int lane   = tid & 31;
  const int wave   = tid >> 5;
  const int gate   = wave & 3;
  const int rowsub = wave >> 2;
  const int n0 = blockIdx.x * 16;   // hidden-col tile
  const int r0 = blockIdx.y * 32;   // batch-row tile

  v8f acc = {};

  // A staging: 32x64 = 2048 elems / 256 thr = 8 consecutive k each
  const int arow = tid >> 3;
  const int ak0  = (tid & 7) * 8;
  // B staging: 4x16x64 = 4096 elems / 256 thr = 16 k each (2x 16B chunks)
  const int be0  = tid * 16;
  const int bg   = be0 >> 10;          // gate 0..3
  const int bn   = (be0 >> 6) & 15;    // local col
  const int bk0  = be0 & 63;           // multiple of 16
  const int bcol = bg * 512 + n0 + bn; // global gate column in [0,2048)
  const uint32_t bs_off = lds_off32(&Bs[(bg * 16 + bn) * 64 + bk0]);

  for (int kk = 0; kk < KTOT; kk += 64) {
    // Async B-tile copy: bf16 weights, contiguous in k (gates = h @ W^T,
    // row-major W => per-column k contiguous). 16B chunks never straddle
    // the 128-boundary (all offsets multiples of 8).
#pragma unroll
    for (int c = 0; c < 2; ++c) {
      int kg = kk + bk0 + c * 8;
      const bf16_t* src = (kg < 128) ? (Wih + bcol * 128 + kg)
                                     : (Whh + bcol * 512 + (kg - 128));
      async_copy_b128(bs_off + c * 16, src);
    }
    // VALU A-tile staging with fp32->bf16 conversion.
#pragma unroll
    for (int j = 0; j < 8; ++j) {
      int kg = kk + ak0 + j;
      float v;
      if (kg < 64)        v = a_t[(r0 + arow) * 64 + kg];
      else if (kg < 128)  v = tval;  // time feature channel
      else                v = (float)hsrc[(r0 + arow) * LSZ + (kg - 128)];
      As[arow * 64 + ak0 + j] = (bf16_t)v;
    }
    // Prefetch next h chunk on the serial critical path.
    if (kk + 64 < KTOT && kk + 64 >= 128)
      __builtin_prefetch(hsrc + (r0 + arow) * LSZ + (kk + 64 - 128) + ak0, 0, 1);

    wait_async0();        // own async copies landed in LDS
    __syncthreads();      // + everyone's ds_stores visible

    const int kbase = (lane & 16) ? 8 : 0;
    const bf16_t* arow_base = &As[(rowsub * 16 + (lane & 15)) * 64];
    const bf16_t* brow_base = &Bs[(gate * 16 + (lane & 15)) * 64];
#pragma unroll
    for (int s = 0; s < 2; ++s) {
      v16bf af = load_frag(arow_base + s * 32, kbase);
      v16bf bf = load_frag(brow_base + s * 32, kbase);
      acc = wmma_bf16(af, bf, acc);
    }
    __syncthreads();
  }

  // Spill gate accumulators to LDS (C/D layout: vgpr v, lanes<16 -> M=v,
  // lanes>=16 -> M=v+8, N = lane&15).
  {
    const int nloc = lane & 15;
    const int mhi  = (lane >> 4) * 8;
#pragma unroll
    for (int v = 0; v < 8; ++v)
      Gs[(gate * 32 + rowsub * 16 + mhi + v) * 16 + nloc] = acc[v];
  }
  __syncthreads();

  // Fused LSTM elementwise update: 512 outputs, 2 per thread.
#pragma unroll
  for (int e2 = 0; e2 < 2; ++e2) {
    int e = tid * 2 + e2;
    int m = e >> 4, n = e & 15;
    int b = r0 + m, nn = n0 + n;
    float gi = Gs[(0 * 32 + m) * 16 + n] + b_ih[nn]         + b_hh[nn];
    float gf = Gs[(1 * 32 + m) * 16 + n] + b_ih[512 + nn]   + b_hh[512 + nn];
    float gg = Gs[(2 * 32 + m) * 16 + n] + b_ih[1024 + nn]  + b_hh[1024 + nn];
    float go = Gs[(3 * 32 + m) * 16 + n] + b_ih[1536 + nn]  + b_hh[1536 + nn];
    float cold = cstate[b * LSZ + nn];
    float cnew = sigmoidf_(gf) * cold + sigmoidf_(gi) * tanhf(gg);
    float hnew = sigmoidf_(go) * tanhf(cnew);
    cstate[b * LSZ + nn]           = cnew;
    hs_out_t[(size_t)b * LSZ + nn] = hnew;
    hdst[b * LSZ + nn]             = (bf16_t)hnew;
  }
}

// ---------------------------------------------------------------------------
// Kernel 2: E = ELU(hs @ W1^T + b1), cols padded 513 -> 576 (zeros).
// grid = (NPAD/64, (T*B)/64), block = 256. 64x64 tile, 2 accs/wave.
// A (fp32 hs) staged via VALU convert; B (bf16 W1) via async-to-LDS.
// ---------------------------------------------------------------------------
__global__ __launch_bounds__(256) void head1_kernel(
    const float*  __restrict__ hs,    // (51200, 512)
    const bf16_t* __restrict__ W1bf,  // (513, 512)
    const float*  __restrict__ b1,    // 513
    bf16_t*       __restrict__ E)     // (51200, NPAD)
{
  __shared__ __align__(16) bf16_t As[64 * 32];
  __shared__ __align__(16) bf16_t Bs[64 * 32];

  const int tid  = threadIdx.x;
  const int lane = tid & 31;
  const int wave = tid >> 5;
  const int rw = wave >> 1;   // row subtile 0..3
  const int cw = wave & 1;    // col half
  const int n0 = blockIdx.x * 64;
  const int m0 = blockIdx.y * 64;

  v8f acc0 = {}, acc1 = {};
  const int e0   = tid * 8;
  const int srow = e0 >> 5;
  const int sk0  = e0 & 31;
  const int ng   = n0 + srow;  // global W1 row for B staging
  const uint32_t bs_off = lds_off32(&Bs[srow * 32 + sk0]);

  for (int kk = 0; kk < 512; kk += 32) {
    if (ng < 513) {
      async_copy_b128(bs_off, W1bf + ng * 512 + kk + sk0);
    } else {
      v8bf z = {};
      *reinterpret_cast<v8bf*>(&Bs[srow * 32 + sk0]) = z;
    }
#pragma unroll
    for (int j = 0; j < 8; ++j)
      As[srow * 32 + sk0 + j] =
          (bf16_t)hs[(size_t)(m0 + srow) * 512 + kk + sk0 + j];
    wait_async0();
    __syncthreads();
    const int kbase = (lane & 16) ? 8 : 0;
    v16bf af  = load_frag(&As[(rw * 16 + (lane & 15)) * 32], kbase);
    v16bf bf0 = load_frag(&Bs[(cw * 32 + (lane & 15)) * 32], kbase);
    v16bf bf1 = load_frag(&Bs[(cw * 32 + 16 + (lane & 15)) * 32], kbase);
    acc0 = wmma_bf16(af, bf0, acc0);
    acc1 = wmma_bf16(af, bf1, acc1);
    __syncthreads();
  }

  const int nloc = lane & 15;
  const int mhi  = (lane >> 4) * 8;
#pragma unroll
  for (int v = 0; v < 8; ++v) {
    int m  = m0 + rw * 16 + mhi + v;
    int na = n0 + cw * 32 + nloc;
    int nb = na + 16;
    float za = 0.0f, zb = 0.0f;
    if (na < 513) { za = acc0[v] + b1[na]; za = (za > 0.f) ? za : (__expf(za) - 1.f); }
    if (nb < 513) { zb = acc1[v] + b1[nb]; zb = (zb > 0.f) ? zb : (__expf(zb) - 1.f); }
    E[(size_t)m * NPAD + na] = (bf16_t)za;
    E[(size_t)m * NPAD + nb] = (bf16_t)zb;
  }
}

// ---------------------------------------------------------------------------
// Kernel 3: x_hat = E @ W2^T + b2. grid = ((T*B)/64), block = 256.
// Both tiles are bf16 copies -> both staged with async-to-LDS.
// ---------------------------------------------------------------------------
__global__ __launch_bounds__(256) void head2_kernel(
    const bf16_t* __restrict__ E,     // (51200, NPAD)
    const bf16_t* __restrict__ W2bf,  // (64, NPAD)
    const float*  __restrict__ b2,    // 64
    float*        __restrict__ out)   // (51200, 64)
{
  __shared__ __align__(16) bf16_t As[64 * 32];
  __shared__ __align__(16) bf16_t Bs[64 * 32];

  const int tid  = threadIdx.x;
  const int lane = tid & 31;
  const int wave = tid >> 5;
  const int rw = wave >> 1;
  const int cw = wave & 1;
  const int m0 = blockIdx.x * 64;

  v8f acc0 = {}, acc1 = {};
  const int e0   = tid * 8;
  const int srow = e0 >> 5;
  const int sk0  = e0 & 31;
  const uint32_t as_off = lds_off32(&As[srow * 32 + sk0]);
  const uint32_t bs_off = lds_off32(&Bs[srow * 32 + sk0]);

  for (int kk = 0; kk < NPAD; kk += 32) {
    async_copy_b128(as_off, E + (size_t)(m0 + srow) * NPAD + kk + sk0);
    async_copy_b128(bs_off, W2bf + srow * NPAD + kk + sk0);
    wait_async0();
    __syncthreads();
    const int kbase = (lane & 16) ? 8 : 0;
    v16bf af  = load_frag(&As[(rw * 16 + (lane & 15)) * 32], kbase);
    v16bf bf0 = load_frag(&Bs[(cw * 32 + (lane & 15)) * 32], kbase);
    v16bf bf1 = load_frag(&Bs[(cw * 32 + 16 + (lane & 15)) * 32], kbase);
    acc0 = wmma_bf16(af, bf0, acc0);
    acc1 = wmma_bf16(af, bf1, acc1);
    __syncthreads();
  }

  const int nloc = lane & 15;
  const int mhi  = (lane >> 4) * 8;
#pragma unroll
  for (int v = 0; v < 8; ++v) {
    int m  = m0 + rw * 16 + mhi + v;
    int na = cw * 32 + nloc;
    int nb = na + 16;
    out[(size_t)m * 64 + na] = acc0[v] + b2[na];
    out[(size_t)m * 64 + nb] = acc1[v] + b2[nb];
  }
}

// ---------------------------------------------------------------------------
// Host launcher (graph-capture safe: only kernel launches on `stream`).
// Inputs: init, a, s, W_ih, W_hh, b_ih, b_hh, W1, b1, W2, b2
// d_out = [x_hat (T*B*64 f32) | hs (T*B*512 f32)]
// ---------------------------------------------------------------------------
extern "C" void kernel_launch(void* const* d_in, const int* in_sizes, int n_in,
                              void* d_out, int out_size, void* d_ws, size_t ws_size,
                              hipStream_t stream) {
  (void)in_sizes; (void)n_in; (void)out_size; (void)ws_size;
  const float* init_ = (const float*)d_in[0];
  const float* a     = (const float*)d_in[1];
  /* d_in[2] = s (unused by reference output path) */
  const float* W_ih  = (const float*)d_in[3];
  const float* W_hh  = (const float*)d_in[4];
  const float* b_ih  = (const float*)d_in[5];
  const float* b_hh  = (const float*)d_in[6];
  const float* W1    = (const float*)d_in[7];
  const float* b1    = (const float*)d_in[8];
  const float* W2    = (const float*)d_in[9];
  const float* b2    = (const float*)d_in[10];

  float* xhat = (float*)d_out;
  float* hs   = xhat + (size_t)TSZ * BSZ * 64;   // hs section of d_out

  // Workspace layout (bf16 weights stay resident in the 192MB L2)
  char* ws = (char*)d_ws;
  size_t off = 0;
  auto alignup = [](size_t x) { return (x + 255) & ~(size_t)255; };
  bf16_t* Wih_bf = (bf16_t*)(ws + off); off = alignup(off + 2048 * 128 * 2);
  bf16_t* Whh_bf = (bf16_t*)(ws + off); off = alignup(off + 2048 * 512 * 2);
  bf16_t* W1_bf  = (bf16_t*)(ws + off); off = alignup(off + 513 * 512 * 2);
  bf16_t* W2_bf  = (bf16_t*)(ws + off); off = alignup(off + 64 * NPAD * 2);
  bf16_t* hbuf0  = (bf16_t*)(ws + off); off = alignup(off + (size_t)BSZ * LSZ * 2);
  bf16_t* hbuf1  = (bf16_t*)(ws + off); off = alignup(off + (size_t)BSZ * LSZ * 2);
  float*  cst    = (float*)(ws + off);  off = alignup(off + (size_t)BSZ * LSZ * 4);
  bf16_t* E      = (bf16_t*)(ws + off); off = alignup(off + (size_t)TSZ * BSZ * NPAD * 2);

  const long Nprep = 2048L * 128 + 2048L * 512 + 513L * 512 + 64L * NPAD +
                     (long)BSZ * LSZ + (long)BSZ * LSZ;
  prep_kernel<<<(int)((Nprep + 255) / 256), 256, 0, stream>>>(
      W_ih, W_hh, W1, W2, init_, Wih_bf, Whh_bf, W1_bf, W2_bf, hbuf0, cst);

  // Sequential recurrence: one fused kernel per timestep, h ping-pong in bf16.
  for (int t = 0; t < TSZ; ++t) {
    const bf16_t* hsrc = (t & 1) ? hbuf1 : hbuf0;
    bf16_t*       hdst = (t & 1) ? hbuf0 : hbuf1;
    lstm_step_kernel<<<dim3(LSZ / 16, BSZ / 32), 256, 0, stream>>>(
        a + (size_t)t * BSZ * 64, hsrc, hdst, cst,
        hs + (size_t)t * BSZ * LSZ,
        Wih_bf, Whh_bf, b_ih, b_hh, (float)t / 100.0f);
  }

  // Output head (fully parallel over T*B = 51200 rows).
  head1_kernel<<<dim3(NPAD / 64, (TSZ * BSZ) / 64), 256, 0, stream>>>(
      hs, W1_bf, b1, E);
  head2_kernel<<<dim3((TSZ * BSZ) / 64), 256, 0, stream>>>(E, W2_bf, b2, xhat);
}
